// additive_attention_layer_4724464025890
// MI455X (gfx1250) — matrix operand order repaired and checked by reading, not compiled, not run on hardware
//
#include <hip/hip_runtime.h>

#define B_ 64
#define T_ 512
#define H_ 1024
#define A_ 256
#define PADVAL (-4194304.0f)

typedef __attribute__((ext_vector_type(16))) __bf16 bf16x16;
typedef __attribute__((ext_vector_type(8)))  float  f32x8;
typedef __attribute__((ext_vector_type(4)))  unsigned int uint4v;

union Frag {
  bf16x16 v;
  uint4v  q[2];
  unsigned int u[8];
  unsigned short h[16];
};

__device__ __forceinline__ unsigned short f2bf(float f) {
  unsigned int u = __float_as_uint(f);
  u += 0x7FFFu + ((u >> 16) & 1u);   // round-to-nearest-even
  return (unsigned short)(u >> 16);
}

#if __has_builtin(__builtin_amdgcn_tanhf)
#define TANH(x) __builtin_amdgcn_tanhf(x)
#else
#define TANH(x) tanhf(x)
#endif

__device__ __forceinline__ f32x8 wmma_bf16(const Frag& a, const Frag& b, f32x8 c) {
  return __builtin_amdgcn_wmma_f32_16x16x32_bf16(false, a.v, false, b.v,
                                                 (short)0, c, false, false);
}

#define SEL_LO 0x01000504u   // { rb.lo16 : ra.lo16 }  (v_perm: src0 = high bytes)
#define SEL_HI 0x03020706u   // { rb.hi16 : ra.hi16 }

// ---------------------------------------------------------------------------
// Kernel 0: convert X, W, U from f32 to bf16 in workspace
// ---------------------------------------------------------------------------
__global__ __launch_bounds__(256) void k_cvt(const float* __restrict__ X,
                                             const float* __restrict__ W,
                                             const float* __restrict__ U,
                                             unsigned short* __restrict__ Xb,
                                             unsigned short* __restrict__ Wb,
                                             unsigned short* __restrict__ Ub) {
  const size_t NX = (size_t)B_ * T_ * H_;
  const size_t NW = (size_t)H_ * A_;
  const size_t NU = (size_t)A_ * T_;
  const size_t stride = (size_t)gridDim.x * blockDim.x;
  const size_t base = (size_t)blockIdx.x * blockDim.x + threadIdx.x;
  for (size_t i = base; i < NX; i += stride) Xb[i] = f2bf(X[i]);
  for (size_t i = base; i < NW; i += stride) Wb[i] = f2bf(W[i]);
  for (size_t i = base; i < NU; i += stride) Ub[i] = f2bf(U[i]);
}

// ---------------------------------------------------------------------------
// Kernel 1: alpha = tanh(X @ W + b)
//   LDS W panel in fragment-major layout: [kblk32][col][kpair(20-padded)]
//   grid = ((B*T)/64, A/64), 128 threads = 4 waves splitting M
// ---------------------------------------------------------------------------
__global__ __launch_bounds__(128) void k_alpha(
    const unsigned short* __restrict__ Xb,
    const unsigned short* __restrict__ Wb,
    const float* __restrict__ bias,
    unsigned short* __restrict__ alphaB) {
  __shared__ unsigned int Wt[H_ / 32][64][20];   // 160 KB

  const int tid  = threadIdx.x;
  const int lane = tid & 31;
  const int wave = tid >> 5;
  const int g    = lane >> 4;
  const int m    = lane & 15;
  const int n0   = blockIdx.y * 64;

  // stage: thread rp holds rows (2rp, 2rp+1); pair-interleave, scatter by column
  for (int rp = tid; rp < H_ / 2; rp += 128) {
    const unsigned int* ra = (const unsigned int*)(Wb + (size_t)(2 * rp) * A_ + n0);
    const unsigned int* rb = ra + A_ / 2;
    const int kb = rp >> 4, pi = rp & 15;
#pragma unroll
    for (int q2 = 0; q2 < 32; ++q2) {
      Wt[kb][2 * q2][pi]     = __builtin_amdgcn_perm(ra[q2], rb[q2], SEL_LO);
      Wt[kb][2 * q2 + 1][pi] = __builtin_amdgcn_perm(ra[q2], rb[q2], SEL_HI);
    }
  }
  __syncthreads();

  const int row0 = blockIdx.x * 64 + wave * 16;
  const int arow = row0 + m;

  float bv[4];
#pragma unroll
  for (int t4 = 0; t4 < 4; ++t4) bv[t4] = bias[n0 + t4 * 16 + m];

  f32x8 acc[4];
#pragma unroll
  for (int t4 = 0; t4 < 4; ++t4) acc[t4] = (f32x8){0, 0, 0, 0, 0, 0, 0, 0};

  for (int kk = 0; kk < H_ / 32; ++kk) {
    Frag af;
    const uint4v* xp = (const uint4v*)(Xb + (size_t)arow * H_ + kk * 32);
    af.q[0] = xp[g];        // K = g*8 .. g*8+7
    af.q[1] = xp[2 + g];    // K = 16+g*8 .. 16+g*8+7
#pragma unroll
    for (int t4 = 0; t4 < 4; ++t4) {
      Frag bf;
      const unsigned int* col = &Wt[kk][t4 * 16 + m][g * 4];
      bf.q[0] = *(const uint4v*)col;
      bf.q[1] = *(const uint4v*)(col + 8);
      acc[t4] = wmma_bf16(af, bf, acc[t4]);
    }
  }

#pragma unroll
  for (int t4 = 0; t4 < 4; ++t4) {
    const int col = n0 + t4 * 16 + m;
#pragma unroll
    for (int r = 0; r < 8; ++r) {
      const int row = row0 + r + 8 * g;
      alphaB[(size_t)row * A_ + col] = f2bf(TANH(acc[t4][r] + bv[t4]));
    }
  }
}

// ---------------------------------------------------------------------------
// staging helpers for k_attn (fragment-major, pair-interleaved, 20-dw stride)
// ---------------------------------------------------------------------------
__device__ __forceinline__ void stage_u(unsigned int (*Utb)[16][20],
                                        const unsigned short* __restrict__ Ub,
                                        int n0, int tid) {
  const int rp = tid;                               // 128 kpairs of the U slice
  const unsigned int* ra = (const unsigned int*)(Ub + (size_t)(2 * rp) * T_ + n0);
  const unsigned int* rb = ra + T_ / 2;
  const int kb = rp >> 4, pi = rp & 15;
#pragma unroll
  for (int q2 = 0; q2 < 8; ++q2) {
    Utb[kb][2 * q2][pi]     = __builtin_amdgcn_perm(ra[q2], rb[q2], SEL_LO);
    Utb[kb][2 * q2 + 1][pi] = __builtin_amdgcn_perm(ra[q2], rb[q2], SEL_HI);
  }
}

__device__ __forceinline__ void stage_x(unsigned int (*Xtb)[32][20],
                                        const unsigned short* __restrict__ Xb,
                                        int b, int h0, int tid, bool pf) {
#pragma unroll
  for (int rr = 0; rr < 2; ++rr) {
    const int rp = tid + rr * 128;                  // 256 kpairs of the X slice
    const unsigned int* ra =
        (const unsigned int*)(Xb + ((size_t)b * T_ + 2 * rp) * H_ + h0);
    const unsigned int* rb = ra + H_ / 2;
    const int kb = rp >> 4, pi = rp & 15;
#pragma unroll
    for (int q2 = 0; q2 < 16; ++q2) {
      Xtb[kb][2 * q2][pi]     = __builtin_amdgcn_perm(ra[q2], rb[q2], SEL_LO);
      Xtb[kb][2 * q2 + 1][pi] = __builtin_amdgcn_perm(ra[q2], rb[q2], SEL_HI);
    }
    if (pf) __builtin_prefetch((const void*)(ra + 16), 0, 0);
  }
}

// ---------------------------------------------------------------------------
// Kernel 2: fused scores -> masked softmax -> Z = p @ X
//   grid = B*(T/64), 128 threads = 4 waves, each owns a 16-query tile
// ---------------------------------------------------------------------------
__global__ __launch_bounds__(128) void k_attn(
    const unsigned short* __restrict__ Xb,
    const unsigned short* __restrict__ Ub,
    const unsigned short* __restrict__ alphaB,
    const int* __restrict__ sen_len,
    float* __restrict__ out) {
  __shared__ float sc[4][16][T_];                 // 128 KB scores/p
  __shared__ unsigned int Ut[2][A_ / 32][16][20]; // 20 KB  double-buffered U
  __shared__ unsigned int Xt[2][T_ / 32][32][20]; // 80 KB  double-buffered X

  const int tid  = threadIdx.x;
  const int lane = tid & 31;
  const int wave = tid >> 5;
  const int g    = lane >> 4;
  const int m    = lane & 15;

  const int b    = blockIdx.x >> 3;
  const int q0   = (blockIdx.x & 7) * 64;
  const int qrow = q0 + wave * 16;
  const int sen  = sen_len[b];

  // preload this wave's alpha tile (16 x 256) as 8 A-fragments (b128 loads)
  Frag afr[8];
  {
    const uint4v* ap =
        (const uint4v*)(alphaB + ((size_t)b * T_ + qrow + m) * A_);
#pragma unroll
    for (int f = 0; f < 8; ++f) {
      afr[f].q[0] = ap[f * 4 + g];
      afr[f].q[1] = ap[f * 4 + 2 + g];
    }
  }

  // ---- Phase A: scores = alpha @ U (double-buffered key slices) ------------
  stage_u(Ut[0], Ub, 0, tid);
  __syncthreads();
  for (int nt = 0; nt < T_ / 16; ++nt) {
    if (nt + 1 < T_ / 16) stage_u(Ut[(nt + 1) & 1], Ub, (nt + 1) * 16, tid);

    f32x8 acc = (f32x8){0, 0, 0, 0, 0, 0, 0, 0};
#pragma unroll
    for (int f = 0; f < 8; ++f) {
      Frag bf;
      const unsigned int* col = &Ut[nt & 1][f][m][g * 4];
      bf.q[0] = *(const uint4v*)col;
      bf.q[1] = *(const uint4v*)(col + 8);
      acc = wmma_bf16(afr[f], bf, acc);
    }
    const int  key   = nt * 16 + m;
    const bool valid = key < sen;
#pragma unroll
    for (int r = 0; r < 8; ++r)
      sc[wave][r + 8 * g][key] = valid ? acc[r] : PADVAL;
    __syncthreads();
  }

  // ---- masked softmax over keys, in place ---------------------------------
  {
    float* row = &sc[wave][m][0];
    const int c0 = g * 256;
    float mx = -3.0e38f;
    for (int c = 0; c < 256; ++c) mx = fmaxf(mx, row[c0 + c]);
    mx = fmaxf(mx, __shfl_xor(mx, 16, 32));
    float sum = 0.f;
    for (int c = 0; c < 256; ++c) sum += __expf(row[c0 + c] - mx);
    sum += __shfl_xor(sum, 16, 32);
    const float inv = 1.0f / sum;
    for (int c = 0; c < 256; ++c)
      row[c0 + c] = __expf(row[c0 + c] - mx) * inv;
  }
  __syncthreads();

  // ---- convert p once into register-resident bf16 A-fragments -------------
  Frag pfr[16];
  {
    const float* prow = &sc[wave][m][0];
#pragma unroll
    for (int kk = 0; kk < 16; ++kk) {
#pragma unroll
      for (int j = 0; j < 4; ++j) {
        const int ka = kk * 32 + g * 8 + 2 * j;
        pfr[kk].h[2 * j]         = f2bf(prow[ka]);
        pfr[kk].h[2 * j + 1]     = f2bf(prow[ka + 1]);
        pfr[kk].h[8 + 2 * j]     = f2bf(prow[ka + 16]);
        pfr[kk].h[8 + 2 * j + 1] = f2bf(prow[ka + 17]);
      }
    }
  }

  // ---- Phase B: Z = p @ X, 32 H-columns per stage, double-buffered ---------
  float* orow = out + ((size_t)b * T_ + qrow) * H_;
  stage_x(Xt[0], Xb, b, 0, tid, true);
  __syncthreads();
  for (int ht = 0; ht < H_ / 32; ++ht) {
    if (ht + 1 < H_ / 32)
      stage_x(Xt[(ht + 1) & 1], Xb, b, (ht + 1) * 32, tid, ht + 2 < H_ / 32);

    f32x8 acc0 = (f32x8){0, 0, 0, 0, 0, 0, 0, 0};
    f32x8 acc1 = (f32x8){0, 0, 0, 0, 0, 0, 0, 0};
#pragma unroll
    for (int kk = 0; kk < T_ / 32; ++kk) {
      Frag x0, x1;
      const unsigned int* c0p = &Xt[ht & 1][kk][m][g * 4];
      const unsigned int* c1p = &Xt[ht & 1][kk][16 + m][g * 4];
      x0.q[0] = *(const uint4v*)c0p;
      x0.q[1] = *(const uint4v*)(c0p + 8);
      x1.q[0] = *(const uint4v*)c1p;
      x1.q[1] = *(const uint4v*)(c1p + 8);
      acc0 = wmma_bf16(pfr[kk], x0, acc0);
      acc1 = wmma_bf16(pfr[kk], x1, acc1);
    }
    const int h0 = ht * 32;
#pragma unroll
    for (int r = 0; r < 8; ++r) {
      orow[(size_t)(r + 8 * g) * H_ + h0 + m]      = acc0[r];
      orow[(size_t)(r + 8 * g) * H_ + h0 + 16 + m] = acc1[r];
    }
    __syncthreads();
  }
}

// ---------------------------------------------------------------------------
extern "C" void kernel_launch(void* const* d_in, const int* in_sizes, int n_in,
                              void* d_out, int out_size, void* d_ws, size_t ws_size,
                              hipStream_t stream) {
  (void)in_sizes; (void)n_in; (void)out_size;
  const float* X    = (const float*)d_in[0];
  const int*   sen  = (const int*)d_in[1];
  const float* W    = (const float*)d_in[2];
  const float* bias = (const float*)d_in[3];
  const float* U    = (const float*)d_in[4];
  float* out = (float*)d_out;

  unsigned short* Xb     = (unsigned short*)d_ws;
  unsigned short* Wb     = Xb + (size_t)B_ * T_ * H_;
  unsigned short* Ubf    = Wb + (size_t)H_ * A_;
  unsigned short* alphaB = Ubf + (size_t)A_ * T_;
  const size_t need = ((size_t)B_ * T_ * H_ + (size_t)H_ * A_ +
                       (size_t)A_ * T_ + (size_t)B_ * T_ * A_) * sizeof(unsigned short);
  if (ws_size < need) return;

  k_cvt<<<2048, 256, 0, stream>>>(X, W, U, Xb, Wb, Ubf);
  k_alpha<<<dim3((B_ * T_) / 64, A_ / 64), 128, 0, stream>>>(Xb, Wb, bias, alphaB);
  k_attn<<<B_ * (T_ / 64), 128, 0, stream>>>(Xb, Ubf, alphaB, sen, out);
}
